// LSTMAutoEncoder_12403865551682
// MI455X (gfx1250) — matrix operand order repaired and checked by reading, not compiled, not run on hardware
//
#include <hip/hip_runtime.h>
#include <hip/hip_bf16.h>

typedef __attribute__((ext_vector_type(16))) _Float16 v16h;
typedef __attribute__((ext_vector_type(8)))  float    v8f;

#define H16   16
#define D32   32
#define T512  512
#define WAVES 8

static __device__ __forceinline__ v8f wmma16(v16h a, v16h b, v8f c) {
  // D = A(16x32 f16) x B(32x16 f16) + C(16x16 f32)
  return __builtin_amdgcn_wmma_f32_16x16x32_f16(false, a, false, b, (short)0, c,
                                                false, false);
}
// Same op, but hint HW to cache the A operand for the next identical WMMA.
static __device__ __forceinline__ v8f wmma16_ra(v16h a, v16h b, v8f c) {
  return __builtin_amdgcn_wmma_f32_16x16x32_f16(false, a, false, b, (short)0, c,
                                                true, false);
}
static __device__ __forceinline__ float fsigmoid(float x) {
  return 1.0f / (1.0f + __expf(-x));
}
static __device__ __forceinline__ float ftanh(float x) {
  x = fminf(fmaxf(x, -15.0f), 15.0f);
  float e = __expf(2.0f * x);
  return (e - 1.0f) / (e + 1.0f);
}
static __device__ __forceinline__ float fleaky(float x) {
  return x >= 0.0f ? x : 0.01f * x;
}

// ---------------------------------------------------------------------------
// Phase 1: x = (leaky(leaky(seq@W1^T+b1)@W2^T+b2))@W3^T+b3  -> f16 (B*T, 16)
// One wave per 16-row tile; seq tile is a native 16x32 A-fragment.
// seq is a one-touch 134MB stream -> non-temporal loads (keep L2 for x/out).
// ---------------------------------------------------------------------------
__global__ __launch_bounds__(256) void pre_mlp_wmma(
    const float* __restrict__ seq,
    const float* __restrict__ w1, const float* __restrict__ b1,
    const float* __restrict__ w2, const float* __restrict__ b2,
    const float* __restrict__ w3, const float* __restrict__ b3,
    _Float16* __restrict__ xout, int nrows)
{
  __shared__ _Float16 stage[WAVES][16 * 16];
  const int lane = threadIdx.x & 31;
  const int wave = threadIdx.x >> 5;
  const int m    = lane & 15;       // A row / C row base
  const int n    = lane & 15;       // B column / C column
  const int hi   = lane >> 4;       // half-wave selector
  const int row0 = (blockIdx.x * WAVES + wave) * 16;
  if (row0 >= nrows) return;        // wave-uniform

  // B fragments. w1: (16,32) full K=32; w2,w3: (16,16) K-padded to 32.
  v16h bw1, bw2, bw3;
#pragma unroll
  for (int i = 0; i < 16; ++i) bw1[i] = (_Float16)w1[n * 32 + (i < 8 ? hi * 8 + i : 16 + hi * 8 + (i - 8))];
#pragma unroll
  for (int i = 0; i < 8; ++i) {
    bw2[i] = (_Float16)w2[n * 16 + hi * 8 + i];
    bw3[i] = (_Float16)w3[n * 16 + hi * 8 + i];
    bw2[8 + i] = (_Float16)0.0f; bw3[8 + i] = (_Float16)0.0f;
  }
  const float bb1 = b1[n], bb2 = b2[n], bb3 = b3[n];

  // A fragment from f32 seq tile (16x32 row-major), streamed non-temporally.
  v16h a;
  const float* srow = seq + (size_t)(row0 + m) * D32;
#pragma unroll
  for (int i = 0; i < 8; ++i) {
    a[i]     = (_Float16)__builtin_nontemporal_load(srow + hi * 8 + i);
    a[8 + i] = (_Float16)__builtin_nontemporal_load(srow + 16 + hi * 8 + i);
  }

  v8f zero = {};
  _Float16* st = stage[wave];

  v8f y = wmma16(a, bw1, zero);
#pragma unroll
  for (int j = 0; j < 8; ++j)
    st[(hi * 8 + j) * 16 + n] = (_Float16)fleaky(y[j] + bb1);

  v16h a2;
#pragma unroll
  for (int i = 0; i < 8; ++i) { a2[i] = st[m * 16 + hi * 8 + i]; a2[8 + i] = (_Float16)0.0f; }
  y = wmma16(a2, bw2, zero);
#pragma unroll
  for (int j = 0; j < 8; ++j)
    st[(hi * 8 + j) * 16 + n] = (_Float16)fleaky(y[j] + bb2);

#pragma unroll
  for (int i = 0; i < 8; ++i) { a2[i] = st[m * 16 + hi * 8 + i]; a2[8 + i] = (_Float16)0.0f; }
  y = wmma16(a2, bw3, zero);
#pragma unroll
  for (int j = 0; j < 8; ++j)
    xout[(size_t)(row0 + hi * 8 + j) * H16 + n] = (_Float16)(y[j] + bb3);
}

// ---------------------------------------------------------------------------
// Phase 2: encoder (512 steps) + decoder (512 steps). One wave per 16-batch
// tile; gate preact = [x_t | h] (16x32) @ [Wih_g^T ; Whh_g^T] (32x16): one
// WMMA per gate with A-reuse hints along the 4-gate chain. Weights live in
// VGPR B-fragments for the whole loop. Output is a one-touch 268MB stream ->
// non-temporal stores.
// ---------------------------------------------------------------------------
__global__ __launch_bounds__(256) void lstm_seq_wmma(
    const _Float16* __restrict__ x,       // (B*T, 16) f16
    const float* __restrict__ Wih, const float* __restrict__ Whh,
    const float* __restrict__ bih, const float* __restrict__ bhh,
    const float* __restrict__ l1w, const float* __restrict__ l1b,
    const float* __restrict__ l2w, const float* __restrict__ l2b,
    const float* __restrict__ pw,  const float* __restrict__ pb,
    float* __restrict__ out, int nbatch)
{
  __shared__ _Float16 hbuf[WAVES][256];
  __shared__ _Float16 sbuf[WAVES][256];
  __shared__ _Float16 tbuf[WAVES][256];
  const int lane = threadIdx.x & 31;
  const int wave = threadIdx.x >> 5;
  const int m = lane & 15, n = lane & 15, hi = lane >> 4;
  const int b0 = (blockIdx.x * WAVES + wave) * 16;
  if (b0 >= nbatch) return;             // wave-uniform

  _Float16* hb = hbuf[wave];
  _Float16* sb = sbuf[wave];
  _Float16* tb = tbuf[wave];

  // Gate B fragments: rows K 0..15 = Wih_g^T (elements 0-7), K 16..31 = Whh_g^T (elements 8-15).
  v16h bg0, bg1, bg2, bg3;
  {
#pragma unroll
    for (int i = 0; i < 8; ++i) {
      bg0[i] = (_Float16)Wih[(0 * 16 + n) * 16 + hi * 8 + i];
      bg1[i] = (_Float16)Wih[(1 * 16 + n) * 16 + hi * 8 + i];
      bg2[i] = (_Float16)Wih[(2 * 16 + n) * 16 + hi * 8 + i];
      bg3[i] = (_Float16)Wih[(3 * 16 + n) * 16 + hi * 8 + i];
      bg0[8 + i] = (_Float16)Whh[(0 * 16 + n) * 16 + hi * 8 + i];
      bg1[8 + i] = (_Float16)Whh[(1 * 16 + n) * 16 + hi * 8 + i];
      bg2[8 + i] = (_Float16)Whh[(2 * 16 + n) * 16 + hi * 8 + i];
      bg3[8 + i] = (_Float16)Whh[(3 * 16 + n) * 16 + hi * 8 + i];
    }
  }
  const float gb0 = bih[0 * 16 + n] + bhh[0 * 16 + n];
  const float gb1 = bih[1 * 16 + n] + bhh[1 * 16 + n];
  const float gb2 = bih[2 * 16 + n] + bhh[2 * 16 + n];
  const float gb3 = bih[3 * 16 + n] + bhh[3 * 16 + n];

  // K-padded B fragments for h2l MLP and post projection (N split in two).
  v16h bl1, bl2, bp0, bp1;
#pragma unroll
  for (int i = 0; i < 8; ++i) {
    bl1[i] = (_Float16)l1w[n * 16 + hi * 8 + i];
    bl2[i] = (_Float16)l2w[n * 16 + hi * 8 + i];
    bp0[i] = (_Float16)pw[n * 16 + hi * 8 + i];
    bp1[i] = (_Float16)pw[(16 + n) * 16 + hi * 8 + i];
    bl1[8 + i] = (_Float16)0.0f; bl2[8 + i] = (_Float16)0.0f;
    bp0[8 + i] = (_Float16)0.0f; bp1[8 + i] = (_Float16)0.0f;
  }
  const float lb1 = l1b[n], lb2 = l2b[n];
  const float pbias0 = pb[n], pbias1 = pb[16 + n];

  v8f c = {};
  v8f hc = {};
#pragma unroll
  for (int j = 0; j < 8; ++j) hb[(hi * 8 + j) * 16 + n] = (_Float16)0.0f;

  const _Float16* xrow = x + (size_t)(b0 + m) * T512 * H16 + hi * 8;
  v8f zero = {};

  // ---------------- encoder ----------------
  for (int t = 0; t < T512; ++t) {
    __builtin_prefetch(xrow + 16 * 8, 0, 0);   // global_prefetch_b8, 8 steps ahead
    v16h a;
#pragma unroll
    for (int i = 0; i < 8; ++i) a[i] = xrow[i];                       // x_t half
#pragma unroll
    for (int i = 0; i < 8; ++i) a[8 + i] = hb[m * 16 + hi * 8 + i];   // h half
    v8f gi = wmma16_ra(a, bg0, zero);
    v8f gf = wmma16_ra(a, bg1, zero);
    v8f gg = wmma16_ra(a, bg2, zero);
    v8f go = wmma16   (a, bg3, zero);
#pragma unroll
    for (int j = 0; j < 8; ++j) {
      float iv = fsigmoid(gi[j] + gb0);
      float fv = fsigmoid(gf[j] + gb1);
      float gv = ftanh(gg[j] + gb2);
      float ov = fsigmoid(go[j] + gb3);
      c[j]  = fv * c[j] + iv * gv;
      hc[j] = ov * ftanh(c[j]);
    }
#pragma unroll
    for (int j = 0; j < 8; ++j) hb[(hi * 8 + j) * 16 + n] = (_Float16)hc[j];
    xrow += H16;
  }

  // start0 = x[:, T-1, :]
  {
    const _Float16* xl = x + ((size_t)(b0 + m) * T512 + (T512 - 1)) * H16 + hi * 8;
#pragma unroll
    for (int i = 0; i < 8; ++i) sb[m * 16 + hi * 8 + i] = xl[i];
  }

  const size_t outHalf = (size_t)nbatch * T512 * D32;

  // ---------------- decoder ----------------
  for (int s = 0; s < T512; ++s) {
    v16h a;
#pragma unroll
    for (int i = 0; i < 8; ++i) a[i] = sb[m * 16 + hi * 8 + i];       // start half
#pragma unroll
    for (int i = 0; i < 8; ++i) a[8 + i] = hb[m * 16 + hi * 8 + i];   // h half
    v8f gi = wmma16_ra(a, bg0, zero);
    v8f gf = wmma16_ra(a, bg1, zero);
    v8f gg = wmma16_ra(a, bg2, zero);
    v8f go = wmma16   (a, bg3, zero);
#pragma unroll
    for (int j = 0; j < 8; ++j) {
      float iv = fsigmoid(gi[j] + gb0);
      float fv = fsigmoid(gf[j] + gb1);
      float gv = ftanh(gg[j] + gb2);
      float ov = fsigmoid(go[j] + gb3);
      c[j]  = fv * c[j] + iv * gv;
      hc[j] = ov * ftanh(c[j]);
    }
#pragma unroll
    for (int j = 0; j < 8; ++j) hb[(hi * 8 + j) * 16 + n] = (_Float16)hc[j];

    // A_h = [h | 0] (K padded)
    v16h ah;
#pragma unroll
    for (int i = 0; i < 8; ++i) { ah[i] = hb[m * 16 + hi * 8 + i]; ah[8 + i] = (_Float16)0.0f; }

    // Three consumers of ah back-to-back (A-reuse chain):
    v8f u  = wmma16_ra(ah, bl1, zero);   // h2l layer 1
    v8f o0 = wmma16_ra(ah, bp0, zero);   // post projection N 0..15
    v8f o1 = wmma16   (ah, bp1, zero);   // post projection N 16..31

    // start = leaky(u + lb1) @ l2w^T + lb2
#pragma unroll
    for (int j = 0; j < 8; ++j) tb[(hi * 8 + j) * 16 + n] = (_Float16)fleaky(u[j] + lb1);
    v16h at;
#pragma unroll
    for (int i = 0; i < 8; ++i) { at[i] = tb[m * 16 + hi * 8 + i]; at[8 + i] = (_Float16)0.0f; }
    v8f s2 = wmma16(at, bl2, zero);
#pragma unroll
    for (int j = 0; j < 8; ++j) sb[(hi * 8 + j) * 16 + n] = (_Float16)(s2[j] + lb2);

    // out = h @ pw^T + pb, written to time T-1-s, both tuple copies,
    // as non-temporal one-touch streams.
    const int tt = T512 - 1 - s;
#pragma unroll
    for (int j = 0; j < 8; ++j) {
      const size_t base = ((size_t)(b0 + hi * 8 + j) * T512 + tt) * D32;
      float v0 = o0[j] + pbias0;
      float v1 = o1[j] + pbias1;
      __builtin_nontemporal_store(v0, out + base + n);
      __builtin_nontemporal_store(v1, out + base + 16 + n);
      __builtin_nontemporal_store(v0, out + outHalf + base + n);
      __builtin_nontemporal_store(v1, out + outHalf + base + 16 + n);
    }
  }
}

// ---------------------------------------------------------------------------
extern "C" void kernel_launch(void* const* d_in, const int* in_sizes, int n_in,
                              void* d_out, int out_size, void* d_ws, size_t ws_size,
                              hipStream_t stream) {
  const float* seq    = (const float*)d_in[0];
  const float* pre_w1 = (const float*)d_in[1];
  const float* pre_b1 = (const float*)d_in[2];
  const float* pre_w2 = (const float*)d_in[3];
  const float* pre_b2 = (const float*)d_in[4];
  const float* pre_w3 = (const float*)d_in[5];
  const float* pre_b3 = (const float*)d_in[6];
  const float* encWih = (const float*)d_in[7];
  const float* encWhh = (const float*)d_in[8];
  const float* encbih = (const float*)d_in[9];
  const float* encbhh = (const float*)d_in[10];
  const float* l1w    = (const float*)d_in[11];
  const float* l1b    = (const float*)d_in[12];
  const float* l2w    = (const float*)d_in[13];
  const float* l2b    = (const float*)d_in[14];
  const float* pw     = (const float*)d_in[15];
  const float* pbv    = (const float*)d_in[16];

  const int B = in_sizes[0] / (T512 * D32);
  const int nrows = B * T512;
  _Float16* xbuf = (_Float16*)d_ws;      // (B*T, 16) f16 intermediate

  dim3 blk(32 * WAVES);
  dim3 g1(((nrows / 16) + WAVES - 1) / WAVES);
  pre_mlp_wmma<<<g1, blk, 0, stream>>>(seq, pre_w1, pre_b1, pre_w2, pre_b2,
                                       pre_w3, pre_b3, xbuf, nrows);

  dim3 g2(((B / 16) + WAVES - 1) / WAVES);
  lstm_seq_wmma<<<g2, blk, 0, stream>>>(xbuf, encWih, encWhh, encbih, encbhh,
                                        l1w, l1b, l2w, l2b, pw, pbv,
                                        (float*)d_out, B);
}
